// FlaxLlamaAttention_34016140984755
// MI455X (gfx1250) — matrix operand order repaired and checked by reading, not compiled, not run on hardware
//
#include <hip/hip_runtime.h>

// ---- problem constants (B=1) ----
#define S_LEN  2048
#define DMODEL 1024
#define NHEADS 16
#define HDIM   64

typedef _Float16 h16;
typedef __attribute__((ext_vector_type(16))) _Float16 v16h;
typedef __attribute__((ext_vector_type(8)))  _Float16 v8h;
typedef __attribute__((ext_vector_type(8)))  float    v8f;

// ---------------- device helpers ----------------

__device__ __forceinline__ v8f vzero8() {
    v8f v;
#pragma unroll
    for (int i = 0; i < 8; ++i) v[i] = 0.0f;
    return v;
}

// A-fragment (16x32 f16, row-major source): lane<16 holds K 0..7 & 16..23,
// lane>=16 holds K 8..15 & 24..31.
__device__ __forceinline__ v16h load_a_frag(const h16* base, int lda, int row,
                                            int k0, int lane) {
    const h16* p = base + (size_t)row * lda + k0;
    const int kh = (lane < 16) ? 0 : 8;
    v8h lo = *(const v8h*)(p + kh);
    v8h hi = *(const v8h*)(p + 16 + kh);
    v16h a;
#pragma unroll
    for (int i = 0; i < 8; ++i) { a[i] = lo[i]; a[i + 8] = hi[i]; }
    return a;
}

// B-fragment (32x16 f16): column col contiguous along K in bt[col][k];
// lane<16 -> K 0..7 & 16..23, lane>=16 -> K 8..15 & 24..31. (global, 16B-aligned)
__device__ __forceinline__ v16h load_b_frag(const h16* bt, int ldb, int col,
                                            int k0, int lane) {
    const h16* p = bt + (size_t)col * ldb + k0 + ((lane < 16) ? 0 : 8);
    v8h lo = *(const v8h*)(p);
    v8h hi = *(const v8h*)(p + 16);
    v16h b;
#pragma unroll
    for (int i = 0; i < 8; ++i) { b[i] = lo[i]; b[i + 8] = hi[i]; }
    return b;
}

// Same fragments from an LDS tile with row stride 40 halfs (16B-aligned reads).
#define LDS_STRIDE 40

__device__ __forceinline__ v16h lds_a_frag(const h16* ldsA, int row, int lane) {
    const h16* p = ldsA + row * LDS_STRIDE;
    const int kh = (lane < 16) ? 0 : 8;
    v8h lo = *(const v8h*)(p + kh);
    v8h hi = *(const v8h*)(p + 16 + kh);
    v16h a;
#pragma unroll
    for (int i = 0; i < 8; ++i) { a[i] = lo[i]; a[i + 8] = hi[i]; }
    return a;
}

__device__ __forceinline__ v16h lds_b_frag(const h16* ldsB, int col, int lane) {
    const h16* p = ldsB + col * LDS_STRIDE + ((lane < 16) ? 0 : 8);
    v8h lo = *(const v8h*)(p);
    v8h hi = *(const v8h*)(p + 16);
    v16h b;
#pragma unroll
    for (int i = 0; i < 8; ++i) { b[i] = lo[i]; b[i + 8] = hi[i]; }
    return b;
}

__device__ __forceinline__ v8f wmma16(v16h a, v16h b, v8f c) {
    return __builtin_amdgcn_wmma_f32_16x16x32_f16(
        false, a, false, b, (short)0, c, false, false);
}

// CDNA5 async global->LDS copy (ASYNCcnt-tracked), 16B per lane.
__device__ __forceinline__ void async_b128(unsigned lds_byte_off,
                                           const void* gptr) {
    asm volatile("global_load_async_to_lds_b128 %0, %1, off"
                 :: "v"(lds_byte_off),
                    "v"((unsigned long long)(uintptr_t)gptr)
                 : "memory");
}
__device__ __forceinline__ void wait_async0() {
    asm volatile("s_wait_asynccnt 0x0" ::: "memory");
}

// reductions across the 16-lane half-groups (rows of the C layout)
__device__ __forceinline__ float rmax16(float v) {
#pragma unroll
    for (int off = 8; off >= 1; off >>= 1) v = fmaxf(v, __shfl_xor(v, off, 32));
    return v;
}
__device__ __forceinline__ float rsum16(float v) {
#pragma unroll
    for (int off = 8; off >= 1; off >>= 1) v += __shfl_xor(v, off, 32);
    return v;
}

// ---------------- elementwise kernels ----------------

__global__ void k_cast(const float* __restrict__ x, h16* __restrict__ y, int n) {
    for (int i = blockIdx.x * blockDim.x + threadIdx.x; i < n;
         i += gridDim.x * blockDim.x)
        y[i] = (h16)x[i];
}

// wt[n*D + k] = w[k*D + n]  (cast fp32 -> f16, transpose to N-major)
__global__ void k_transpose_w(const float* __restrict__ w, h16* __restrict__ wt) {
    int tid = blockIdx.x * blockDim.x + threadIdx.x;
    if (tid >= DMODEL * DMODEL) return;
    int ncol = tid / DMODEL;
    int k    = tid - ncol * DMODEL;
    wt[tid] = (h16)w[(size_t)k * DMODEL + ncol];
}

// Interleaved-pair RoPE on Q and K (in place, f16).
__global__ void k_rope(h16* __restrict__ Q, h16* __restrict__ K,
                       const int* __restrict__ pos_ids) {
    int idx = blockIdx.x * blockDim.x + threadIdx.x;   // H*S*(HD/2) = 1M
    if (idx >= NHEADS * S_LEN * (HDIM / 2)) return;
    int i = idx & 31;
    int s = (idx >> 5) & (S_LEN - 1);
    int h = idx >> 16;
    float pos = (float)pos_ids[s];
    float ang = pos * __expf(-(float)i * (9.210340371976184f / 32.0f));
    float c = cosf(ang), sn = sinf(ang);
    size_t base = ((size_t)h * S_LEN + s) * HDIM + 2 * i;

    float q0 = (float)Q[base], q1 = (float)Q[base + 1];
    Q[base]     = (h16)(q0 * c - q1 * sn);
    Q[base + 1] = (h16)(q0 * sn + q1 * c);
    float k0 = (float)K[base], k1 = (float)K[base + 1];
    K[base]     = (h16)(k0 * c - k1 * sn);
    K[base + 1] = (h16)(k0 * sn + k1 * c);
}

// ---------------- LDS-tiled GEMM mainloop ----------------
// Block tile: 128(M) x 128(N), K stepped by 32. 8 waves: wave (wm,wn) computes
// a 32x64 sub-tile -> acc[2 msub][4 ntile]. A and B tiles staged to LDS with
// GLOBAL_LOAD_ASYNC_TO_LDS_B128 (4 x b128 per thread per K-step).
__device__ __forceinline__ void gemm_mainloop_128(
    const h16* __restrict__ A, const h16* __restrict__ Bt,
    int m_base, int n_base, h16* ldsA, h16* ldsB, v8f acc[2][4]) {
    const int lane = threadIdx.x & 31;
    const int wid  = threadIdx.x >> 5;
    const int wm = (wid & 3) * 32;
    const int wn = (wid >> 2) * 64;
    const int n_l = lane & 15;

    // staging mapping: thread covers one 32B (16-half) segment of A and of B
    const int srow = threadIdx.x >> 1;            // 0..127
    const int sseg = (threadIdx.x & 1) * 16;      // halfs
    const h16* gA = A  + (size_t)(m_base + srow) * DMODEL + sseg;
    const h16* gB = Bt + (size_t)(n_base + srow) * DMODEL + sseg;
    const unsigned lA = (unsigned)(uintptr_t)(ldsA + srow * LDS_STRIDE + sseg);
    const unsigned lB = (unsigned)(uintptr_t)(ldsB + srow * LDS_STRIDE + sseg);

#pragma unroll
    for (int ms = 0; ms < 2; ++ms)
#pragma unroll
        for (int t = 0; t < 4; ++t) acc[ms][t] = vzero8();

    for (int k0 = 0; k0 < DMODEL; k0 += 32) {
        async_b128(lA,      gA + k0);
        async_b128(lA + 16, gA + k0 + 8);
        async_b128(lB,      gB + k0);
        async_b128(lB + 16, gB + k0 + 8);
        wait_async0();
        __syncthreads();

        v16h a0 = lds_a_frag(ldsA, wm + n_l, lane);
        v16h a1 = lds_a_frag(ldsA, wm + 16 + n_l, lane);
#pragma unroll
        for (int t = 0; t < 4; ++t) {
            v16h b = lds_b_frag(ldsB, wn + t * 16 + n_l, lane);
            acc[0][t] = wmma16(a0, b, acc[0][t]);
            acc[1][t] = wmma16(a1, b, acc[1][t]);
        }
        __syncthreads();   // protect LDS tile before next stage overwrites
    }
}

// Fused QKV projection with LDS-tiled GEMM.
// which=0 -> Q [H][S][HD], which=1 -> K [H][S][HD], which=2 -> V^T [D][S].
__global__ __launch_bounds__(256) void k_qkv(const h16* __restrict__ X,
                                             const h16* __restrict__ wqT,
                                             const h16* __restrict__ wkT,
                                             const h16* __restrict__ wvT,
                                             h16* __restrict__ Q,
                                             h16* __restrict__ K,
                                             h16* __restrict__ Vt) {
    __shared__ h16 lds[2 * 128 * LDS_STRIDE];     // 20 KB
    h16* ldsA = lds;
    h16* ldsB = lds + 128 * LDS_STRIDE;

    const int bid = blockIdx.x;                    // 0..383
    const int which = bid >> 7;                    // 0..2
    const int rem = bid & 127;
    const int m_base = (rem >> 3) * 128;           // 16 m-bands
    const int n_base = (rem & 7) * 128;            // 8 n-bands
    const h16* wt = (which == 0) ? wqT : (which == 1) ? wkT : wvT;

    v8f acc[2][4];
    gemm_mainloop_128(X, wt, m_base, n_base, ldsA, ldsB, acc);

    const int lane = threadIdx.x & 31;
    const int wid  = threadIdx.x >> 5;
    const int wm = (wid & 3) * 32;
    const int wn = (wid >> 2) * 64;
    const int n_l  = lane & 15;
    const int moff = (lane < 16) ? 0 : 8;

    if (which < 2) {
        h16* dst = (which == 0) ? Q : K;
#pragma unroll
        for (int ms = 0; ms < 2; ++ms)
#pragma unroll
            for (int t = 0; t < 4; ++t) {
                const int col  = n_base + wn + t * 16 + n_l;
                const int head = col >> 6, hd = col & 63;
#pragma unroll
                for (int r = 0; r < 8; ++r) {
                    const int m = m_base + wm + ms * 16 + r + moff;
                    dst[((size_t)head * S_LEN + m) * HDIM + hd] =
                        (h16)acc[ms][t][r];
                }
            }
    } else {
#pragma unroll
        for (int ms = 0; ms < 2; ++ms)
#pragma unroll
            for (int t = 0; t < 4; ++t) {
                const int col = n_base + wn + t * 16 + n_l;
#pragma unroll
                for (int r = 0; r < 8; ++r) {
                    const int m = m_base + wm + ms * 16 + r + moff;
                    Vt[(size_t)col * S_LEN + m] = (h16)acc[ms][t][r];
                }
            }
    }
}

// Output projection: AO[S][D] (f16) x WoT -> fp32 out, LDS-tiled.
__global__ __launch_bounds__(256) void k_oproj(const h16* __restrict__ AO,
                                               const h16* __restrict__ woT,
                                               float* __restrict__ out) {
    __shared__ h16 lds[2 * 128 * LDS_STRIDE];
    h16* ldsA = lds;
    h16* ldsB = lds + 128 * LDS_STRIDE;

    const int bid = blockIdx.x;                    // 0..127
    const int m_base = (bid >> 3) * 128;
    const int n_base = (bid & 7) * 128;

    v8f acc[2][4];
    gemm_mainloop_128(AO, woT, m_base, n_base, ldsA, ldsB, acc);

    const int lane = threadIdx.x & 31;
    const int wid  = threadIdx.x >> 5;
    const int wm = (wid & 3) * 32;
    const int wn = (wid >> 2) * 64;
    const int n_l  = lane & 15;
    const int moff = (lane < 16) ? 0 : 8;

#pragma unroll
    for (int ms = 0; ms < 2; ++ms)
#pragma unroll
        for (int t = 0; t < 4; ++t) {
            const int col = n_base + wn + t * 16 + n_l;
#pragma unroll
            for (int r = 0; r < 8; ++r) {
                const int m = m_base + wm + ms * 16 + r + moff;
                out[(size_t)m * DMODEL + col] = acc[ms][t][r];
            }
        }
}

// ---------------- flash attention (unchanged from round 1) ----------------
__global__ __launch_bounds__(256) void k_attn(const h16* __restrict__ Q,
                                              const h16* __restrict__ K,
                                              const h16* __restrict__ Vt,
                                              const int* __restrict__ amask,
                                              h16* __restrict__ AO) {
    __shared__ h16 plds[8 * 16 * 32];                  // 1KB per wave
    const int lane = threadIdx.x & 31;
    const int wid  = threadIdx.x >> 5;
    const int gw   = blockIdx.x * 8 + wid;             // 0..2047
    const int head = gw >> 7;
    const int q0   = (gw & 127) * 16;
    const int n_l  = lane & 15;
    const int moff = (lane < 16) ? 0 : 8;
    const float scale = 0.125f;                        // 1/sqrt(64)

    const h16* Qh = Q + (size_t)head * S_LEN * HDIM;
    const h16* Kh = K + (size_t)head * S_LEN * HDIM;
    const h16* Vh = Vt + (size_t)head * HDIM * S_LEN;
    h16* myp = plds + wid * 512;

    v16h aq0 = load_a_frag(Qh, HDIM, q0 + n_l, 0, lane);
    v16h aq1 = load_a_frag(Qh, HDIM, q0 + n_l, 32, lane);

    float mrow[8], lrow[8];
#pragma unroll
    for (int r = 0; r < 8; ++r) { mrow[r] = -3.0e38f; lrow[r] = 0.0f; }
    v8f o[4];
#pragma unroll
    for (int t = 0; t < 4; ++t) o[t] = vzero8();

    const int kbmax = (q0 + 15) / 32;
    for (int kb = 0; kb <= kbmax; ++kb) {
        const int k0 = kb * 32;
        v8f s0 = vzero8(), s1 = vzero8();
        {
            v16h b = load_b_frag(Kh, HDIM, k0 + n_l, 0, lane);
            s0 = wmma16(aq0, b, s0);
            b = load_b_frag(Kh, HDIM, k0 + n_l, 32, lane);
            s0 = wmma16(aq1, b, s0);
            b = load_b_frag(Kh, HDIM, k0 + 16 + n_l, 0, lane);
            s1 = wmma16(aq0, b, s1);
            b = load_b_frag(Kh, HDIM, k0 + 16 + n_l, 32, lane);
            s1 = wmma16(aq1, b, s1);
        }
        const int kk0 = k0 + n_l;
        const int kk1 = k0 + 16 + n_l;
        const int ok0 = (amask[kk0] > 0);
        const int ok1 = (amask[kk1] > 0);
#pragma unroll
        for (int r = 0; r < 8; ++r) {
            const int q_idx = q0 + r + moff;
            float v0 = s0[r] * scale;
            float v1 = s1[r] * scale;
            if (kk0 > q_idx || !ok0) v0 = -1.0e30f;
            if (kk1 > q_idx || !ok1) v1 = -1.0e30f;
            float mx = rmax16(fmaxf(v0, v1));
            float mnew = fmaxf(mrow[r], mx);
            float p0 = __expf(v0 - mnew);
            float p1 = __expf(v1 - mnew);
            float sum = rsum16(p0 + p1);
            float corr = __expf(mrow[r] - mnew);
            mrow[r] = mnew;
            lrow[r] = lrow[r] * corr + sum;
#pragma unroll
            for (int t = 0; t < 4; ++t) o[t][r] *= corr;
            const int m = r + moff;
            myp[m * 32 + n_l]      = (h16)p0;
            myp[m * 32 + 16 + n_l] = (h16)p1;
        }
        // per-wave LDS RAW fence (no block barrier: waves diverge in trip count)
        asm volatile("s_wait_dscnt 0x0" ::: "memory");
        {
            const int kh = (lane < 16) ? 0 : 8;
            v8h lo = *(const v8h*)(myp + n_l * 32 + kh);
            v8h hi = *(const v8h*)(myp + n_l * 32 + 16 + kh);
            v16h ap;
#pragma unroll
            for (int i = 0; i < 8; ++i) { ap[i] = lo[i]; ap[i + 8] = hi[i]; }
#pragma unroll
            for (int t = 0; t < 4; ++t) {
                v16h bv = load_b_frag(Vh, S_LEN, t * 16 + n_l, k0, lane);
                o[t] = wmma16(ap, bv, o[t]);
            }
        }
    }

#pragma unroll
    for (int t = 0; t < 4; ++t)
#pragma unroll
        for (int r = 0; r < 8; ++r) {
            float val = o[t][r] / lrow[r];
            AO[(size_t)(q0 + r + moff) * DMODEL + head * HDIM + t * 16 + n_l] =
                (h16)val;
        }
}

// ---------------- host launcher ----------------

extern "C" void kernel_launch(void* const* d_in, const int* in_sizes, int n_in,
                              void* d_out, int out_size, void* d_ws, size_t ws_size,
                              hipStream_t stream) {
    const float* hidden = (const float*)d_in[0];
    const int*   amask  = (const int*)d_in[1];
    const int*   posids = (const int*)d_in[2];
    const float* wq = (const float*)d_in[3];
    const float* wk = (const float*)d_in[4];
    const float* wv = (const float*)d_in[5];
    const float* wo = (const float*)d_in[6];
    float* out = (float*)d_out;

    char* ws = (char*)d_ws;
    h16* Xb  = (h16*)(ws + (size_t)0);           // 4 MB  [S][D]
    h16* WqT = (h16*)(ws + ((size_t)4  << 20));  // 2 MB  [N][K]
    h16* WkT = (h16*)(ws + ((size_t)6  << 20));
    h16* WvT = (h16*)(ws + ((size_t)8  << 20));
    h16* WoT = (h16*)(ws + ((size_t)10 << 20));
    h16* Qb  = (h16*)(ws + ((size_t)12 << 20));  // 4 MB  [H][S][HD]
    h16* Kb  = (h16*)(ws + ((size_t)16 << 20));  // 4 MB  [H][S][HD]
    h16* Vt  = (h16*)(ws + ((size_t)20 << 20));  // 4 MB  [D][S]
    h16* AO  = (h16*)(ws + ((size_t)24 << 20));  // 4 MB  [S][D]

    k_cast<<<2048, 256, 0, stream>>>(hidden, Xb, S_LEN * DMODEL);
    k_transpose_w<<<4096, 256, 0, stream>>>(wq, WqT);
    k_transpose_w<<<4096, 256, 0, stream>>>(wk, WkT);
    k_transpose_w<<<4096, 256, 0, stream>>>(wv, WvT);
    k_transpose_w<<<4096, 256, 0, stream>>>(wo, WoT);

    k_qkv<<<384, 256, 0, stream>>>(Xb, WqT, WkT, WvT, Qb, Kb, Vt);
    k_rope<<<4096, 256, 0, stream>>>(Qb, Kb, posids);
    k_attn<<<256, 256, 0, stream>>>(Qb, Kb, Vt, amask, AO);
    k_oproj<<<128, 256, 0, stream>>>(AO, WoT, out);
}
